// DeepSeekMoE_11115375362382
// MI455X (gfx1250) — matrix operand order repaired and checked by reading, not compiled, not run on hardware
//
#include <hip/hip_runtime.h>
#include <hip/hip_bf16.h>

// Problem constants (from reference): N=2048, H=512, E=8, S=4, F_SH=128
#define NTOK   2048
#define HDIM   512
#define NEXP   8
#define NSUB   4
#define FSH    128
#define NCOMBO 32            // E*S flattened routed combos
#define NGRP   36            // 32 routed + 4 shared groups
#define MT     64            // token rows per FFN workgroup tile

typedef __bf16  bf16_t;
typedef bf16_t  bf16x2  __attribute__((ext_vector_type(2)));
typedef bf16_t  bf16x16 __attribute__((ext_vector_type(16)));
typedef float   v8f     __attribute__((ext_vector_type(8)));

union Frag16 { bf16x16 v; unsigned u[8]; uint4 q[2]; };

// Pack two floats into one dword of bf16 (lo in [15:0], hi in [31:16]).
// Prefer the native packed-convert builtin when declared; otherwise use a
// plain __bf16 cast (fptrunc) and let the gfx1250 backend pick the native
// v_cvt_pk_bf16_f32 / v_cvt_bf16_f32.
__device__ __forceinline__ unsigned pack2_bf16(float lo, float hi) {
#if __has_builtin(__builtin_amdgcn_cvt_pk_bf16_f32)
  union { bf16x2 v; unsigned u; } o;
  o.v = __builtin_amdgcn_cvt_pk_bf16_f32(lo, hi);
  return o.u;
#else
  union { bf16x2 v; unsigned u; } o;
  o.v = bf16x2{(bf16_t)lo, (bf16_t)hi};
  return o.u;
#endif
}

__device__ __forceinline__ bf16_t f2bf(float f) {
#if __has_builtin(__builtin_amdgcn_cvt_pk_bf16_f32)
  bf16x2 v = __builtin_amdgcn_cvt_pk_bf16_f32(f, 0.0f);
  return v[0];
#else
  return (bf16_t)f;
#endif
}

__device__ __forceinline__ float sigmoidf_(float v) { return 1.0f / (1.0f + __expf(-v)); }
__device__ __forceinline__ float gelu_exact(float v) {
  return 0.5f * v * (1.0f + erff(v * 0.70710678118654752f));
}

// ---------------------------------------------------------------------------
// Stage 0: zero the 36 group counters
// ---------------------------------------------------------------------------
__global__ void moe_zero_cnt_kernel(int* __restrict__ cnt) {
  if (threadIdx.x < NGRP) cnt[threadIdx.x] = 0;
}

// ---------------------------------------------------------------------------
// Stage 1: router — one wave32 per token. 16 fused dot products:
//   cols 0..3  = shared_router,  4..7 = sub_router,  8..15 = expert router
// ---------------------------------------------------------------------------
__global__ void moe_router_kernel(
    const float* __restrict__ x,
    const float* __restrict__ shW,  const float* __restrict__ shB,
    const float* __restrict__ subW, const float* __restrict__ subB,
    const float* __restrict__ rW,   const float* __restrict__ rB,
    const float* __restrict__ eBias, const float* __restrict__ seBias,
    int* __restrict__ tc0, float* __restrict__ tw0,
    int* __restrict__ tc1, float* __restrict__ tw1,
    int* __restrict__ tsi, float* __restrict__ tsw,
    int* __restrict__ cnt)
{
  int wave = threadIdx.x >> 5;
  unsigned lane = threadIdx.x & 31u;
  int n = blockIdx.x * (blockDim.x >> 5) + wave;
  if (n >= NTOK) return;

  float acc[16];
#pragma unroll
  for (int j = 0; j < 16; ++j) acc[j] = 0.f;

  for (int k = (int)lane; k < HDIM; k += 32) {
    float  xv = x[(size_t)n * HDIM + k];
    float4 a  = *(const float4*)(shW  + (size_t)k * NSUB);
    float4 b  = *(const float4*)(subW + (size_t)k * NSUB);
    float4 c0 = *(const float4*)(rW   + (size_t)k * NEXP);
    float4 c1 = *(const float4*)(rW   + (size_t)k * NEXP + 4);
    acc[0]  += xv * a.x;  acc[1]  += xv * a.y;  acc[2]  += xv * a.z;  acc[3]  += xv * a.w;
    acc[4]  += xv * b.x;  acc[5]  += xv * b.y;  acc[6]  += xv * b.z;  acc[7]  += xv * b.w;
    acc[8]  += xv * c0.x; acc[9]  += xv * c0.y; acc[10] += xv * c0.z; acc[11] += xv * c0.w;
    acc[12] += xv * c1.x; acc[13] += xv * c1.y; acc[14] += xv * c1.z; acc[15] += xv * c1.w;
  }
#pragma unroll
  for (int j = 0; j < 16; ++j) {
    float v = acc[j];
#pragma unroll
    for (int off = 16; off; off >>= 1) v += __shfl_xor(v, off, 32);
    acc[j] = v;
  }

  if (lane == 0) {
    float sp[NSUB], sub[NSUB], rp[NEXP];
#pragma unroll
    for (int s = 0; s < NSUB; ++s) {
      sp[s]  = sigmoidf_(acc[s]     + shB[s]  + seBias[s]);
      sub[s] = sigmoidf_(acc[4 + s] + subB[s] + seBias[s]);
    }
#pragma unroll
    for (int e = 0; e < NEXP; ++e)
      rp[e] = sigmoidf_(acc[8 + e] + rB[e] + eBias[e]);

    // top-1 shared (stable: strict > keeps lowest index, like jax top_k)
    int si = 0; float sw = sp[0];
#pragma unroll
    for (int s = 1; s < NSUB; ++s) if (sp[s] > sw) { sw = sp[s]; si = s; }
    // top-1 sub-expert index (gate value unused per reference)
    int ssi = 0; float sv = sub[0];
#pragma unroll
    for (int s = 1; s < NSUB; ++s) if (sub[s] > sv) { sv = sub[s]; ssi = s; }
    // top-2 experts
    int e0 = 0; float w0 = rp[0];
#pragma unroll
    for (int e = 1; e < NEXP; ++e) if (rp[e] > w0) { w0 = rp[e]; e0 = e; }
    int e1 = -1; float w1 = -1.0f;
#pragma unroll
    for (int e = 0; e < NEXP; ++e) if (e != e0 && rp[e] > w1) { w1 = rp[e]; e1 = e; }

    int c0 = e0 * NSUB + ssi;
    int c1 = e1 * NSUB + ssi;
    tc0[n] = c0; tw0[n] = w0;
    tc1[n] = c1; tw1[n] = w1;
    tsi[n] = si; tsw[n] = sw;
    atomicAdd(&cnt[c0], 1);
    atomicAdd(&cnt[c1], 1);
    atomicAdd(&cnt[NCOMBO + si], 1);
  }
}

// ---------------------------------------------------------------------------
// Stage 2: exclusive scan over 36 group counts (tiny; single thread)
// ---------------------------------------------------------------------------
__global__ void moe_scan_kernel(const int* __restrict__ cnt,
                                int* __restrict__ offs, int* __restrict__ cursor) {
  if (threadIdx.x == 0) {
    int run = 0;
    for (int g = 0; g < NGRP; ++g) { offs[g] = run; cursor[g] = run; run += cnt[g]; }
    offs[NGRP] = run;
  }
}

// ---------------------------------------------------------------------------
// Stage 3: scatter tokens into per-group lists. Routed entries encode
// (token<<1)|slot so slot0/slot1 results land in disjoint scratch buffers
// (plain stores, bit-deterministic — no float atomics anywhere).
// ---------------------------------------------------------------------------
__global__ void moe_scatter_kernel(
    const int* __restrict__ tc0, const float* __restrict__ tw0,
    const int* __restrict__ tc1, const float* __restrict__ tw1,
    const int* __restrict__ tsi, const float* __restrict__ tsw,
    int* __restrict__ cursor, int* __restrict__ toklist, float* __restrict__ gatelist)
{
  int n = blockIdx.x * blockDim.x + threadIdx.x;
  if (n >= NTOK) return;
  int p0 = atomicAdd(&cursor[tc0[n]], 1);
  toklist[p0] = (n << 1);      gatelist[p0] = tw0[n];
  int p1 = atomicAdd(&cursor[tc1[n]], 1);
  toklist[p1] = (n << 1) | 1;  gatelist[p1] = tw1[n];
  int ps = atomicAdd(&cursor[NCOMBO + tsi[n]], 1);
  toklist[ps] = n;             gatelist[ps] = tsw[n];
}

// ---------------------------------------------------------------------------
// Stage 4: grouped FFN  y = gelu(Xg @ W1 + b1) @ W2 + b2, gate-scaled.
// One workgroup = 64 token rows x all output columns of one group.
// bf16 WMMA 16x16x32, fp32 accumulate. blockDim = 256 (8 wave32s).
// Dynamic LDS: Xs[64][512] bf16 + Hs[64][512] bf16 + toks/gates = 131584 B
// (WGP has 320KB LDS on CDNA5).
// ---------------------------------------------------------------------------
__global__ __launch_bounds__(256, 1)
void moe_ffn_wmma_kernel(
    const float* __restrict__ x,
    const float* __restrict__ W1b, const float* __restrict__ b1b,
    const float* __restrict__ W2b, const float* __restrict__ b2b,
    int F, int groupBase, int routed,
    const int* __restrict__ offs,
    const int* __restrict__ toklist, const float* __restrict__ gates,
    float* __restrict__ dst0, float* __restrict__ dst1)
{
  extern __shared__ char smem_raw[];
  bf16_t* Xs  = (bf16_t*)smem_raw;              // [MT][HDIM] bf16
  bf16_t* Hs  = Xs + (size_t)MT * HDIM;         // [MT][F]    bf16 (F<=HDIM)
  int*   toks = (int*)(Xs + (size_t)2 * MT * HDIM);
  float* gts  = (float*)(toks + MT);

  int gi = blockIdx.x;
  int g  = groupBase + gi;
  int lo = offs[g];
  int cg = offs[g + 1] - lo;
  int m0 = blockIdx.y * MT;
  if (m0 >= cg) return;
  int mcnt = min(MT, cg - m0);

  int tid = threadIdx.x;
  unsigned lane = tid & 31u;
  int wave = tid >> 5;

  const float* W1 = W1b + (size_t)gi * HDIM * F;
  const float* b1 = b1b + (size_t)gi * F;
  const float* W2 = W2b + (size_t)gi * F * HDIM;
  const float* b2 = b2b + (size_t)gi * HDIM;

  if (tid < MT) {
    if (tid < mcnt) {
      int e = toklist[lo + m0 + tid];
      toks[tid] = routed ? e : (e << 1);        // (token<<1)|slot; shared slot=0
      gts[tid]  = gates[lo + m0 + tid];
    } else { toks[tid] = -1; gts[tid] = 0.f; }
  }
  __syncthreads();

  // ---- stage X tile into LDS as bf16 (fully coalesced float4 reads) ----
  for (int r = wave; r < MT; r += 8) {
    int ec = toks[r];
    const float* xr = (ec >= 0) ? (x + (size_t)(ec >> 1) * HDIM) : nullptr;
    for (int cb = 0; cb < HDIM; cb += 128) {
      int c = cb + (int)lane * 4;
      float4 xv = make_float4(0.f, 0.f, 0.f, 0.f);
      if (xr) xv = *(const float4*)(xr + c);
      unsigned p0 = pack2_bf16(xv.x, xv.y);
      unsigned p1 = pack2_bf16(xv.z, xv.w);
      *(uint2*)(Xs + (size_t)r * HDIM + c) = make_uint2(p0, p1);
    }
  }
  __syncthreads();

  // ---- GEMM1: [64,H] @ [H,F] -> gelu -> Hs (bf16) ----
  for (int ct = wave; ct < (F >> 4); ct += 8) {
    int col0 = ct << 4;
    v8f acc[4] = {};
    for (int k0 = 0; k0 < HDIM; k0 += 32) {
      // B fragment: dense 16-bit B layout (lane=col, packed K pairs per VGPR)
      Frag16 bf_;
      {
        int cc = col0 + (int)(lane & 15u);
        int kb = k0 + (int)((lane >> 4) << 4);
        const float* p = W1 + (size_t)kb * F + cc;
#pragma unroll
        for (int vv = 0; vv < 8; ++vv) {
          bf_.u[vv] = pack2_bf16(p[0], p[(size_t)F]);
          p += 2 * (size_t)F;
        }
      }
#pragma unroll
      for (int m = 0; m < 4; ++m) {
        // A fragment: 16-bit A layout == two ds_load_b128 per lane
        Frag16 af;
        int rr = (m << 4) + (int)(lane & 15u);
        int kb = k0 + (int)((lane >> 4) << 4);
        const bf16_t* ap = Xs + (size_t)rr * HDIM + kb;
        af.q[0] = *(const uint4*)ap;         // K = kb..kb+7
        af.q[1] = *(const uint4*)(ap + 8);   // K = kb+8..kb+15
        acc[m] = __builtin_amdgcn_wmma_f32_16x16x32_bf16(
                     false, af.v, false, bf_.v, (short)0, acc[m], false, false);
      }
    }
    float bias = b1[col0 + (int)(lane & 15u)];
#pragma unroll
    for (int m = 0; m < 4; ++m) {
#pragma unroll
      for (int vv = 0; vv < 8; ++vv) {
        int rr = (m << 4) + ((lane < 16u) ? vv : vv + 8);
        float y = gelu_exact(acc[m][vv] + bias);
        Hs[(size_t)rr * F + col0 + (int)(lane & 15u)] = f2bf(y);
      }
    }
  }
  __syncthreads();

  // ---- GEMM2: [64,F] @ [F,H] -> gate-scaled stores ----
  for (int ct = wave; ct < (HDIM >> 4); ct += 8) {
    int col0 = ct << 4;
    v8f acc[4] = {};
    for (int k0 = 0; k0 < F; k0 += 32) {
      Frag16 bf_;
      {
        int cc = col0 + (int)(lane & 15u);
        int kb = k0 + (int)((lane >> 4) << 4);
        const float* p = W2 + (size_t)kb * HDIM + cc;
#pragma unroll
        for (int vv = 0; vv < 8; ++vv) {
          bf_.u[vv] = pack2_bf16(p[0], p[(size_t)HDIM]);
          p += 2 * (size_t)HDIM;
        }
      }
#pragma unroll
      for (int m = 0; m < 4; ++m) {
        Frag16 af;
        int rr = (m << 4) + (int)(lane & 15u);
        int kb = k0 + (int)((lane >> 4) << 4);
        const bf16_t* ap = Hs + (size_t)rr * F + kb;
        af.q[0] = *(const uint4*)ap;
        af.q[1] = *(const uint4*)(ap + 8);
        acc[m] = __builtin_amdgcn_wmma_f32_16x16x32_bf16(
                     false, af.v, false, bf_.v, (short)0, acc[m], false, false);
      }
    }
    int cc = col0 + (int)(lane & 15u);
    float bias = b2[cc];
#pragma unroll
    for (int m = 0; m < 4; ++m) {
#pragma unroll
      for (int vv = 0; vv < 8; ++vv) {
        int rr = (m << 4) + ((lane < 16u) ? vv : vv + 8);
        int ec = toks[rr];
        if (ec >= 0) {
          float val = gts[rr] * (acc[m][vv] + bias);
          float* dst = (ec & 1) ? dst1 : dst0;
          dst[(size_t)(ec >> 1) * HDIM + cc] = val;   // plain store, disjoint
        }
      }
    }
  }
}

// ---------------------------------------------------------------------------
// Stage 5: out = shared_out (already in out) + routed slot0 + slot1
// ---------------------------------------------------------------------------
__global__ void moe_combine_kernel(float* __restrict__ out,
                                   const float* __restrict__ r0,
                                   const float* __restrict__ r1, int n) {
  int i = blockIdx.x * blockDim.x + threadIdx.x;
  if (i < n) out[i] += r0[i] + r1[i];
}

// ---------------------------------------------------------------------------
extern "C" void kernel_launch(void* const* d_in, const int* in_sizes, int n_in,
                              void* d_out, int out_size, void* d_ws, size_t ws_size,
                              hipStream_t stream) {
  (void)in_sizes; (void)n_in; (void)out_size; (void)ws_size;
  const float* x       = (const float*)d_in[0];
  const float* shW1    = (const float*)d_in[1];
  const float* shB1    = (const float*)d_in[2];
  const float* shW2    = (const float*)d_in[3];
  const float* shB2    = (const float*)d_in[4];
  const float* shRW    = (const float*)d_in[5];
  const float* shRB    = (const float*)d_in[6];
  const float* eW1     = (const float*)d_in[7];
  const float* eB1     = (const float*)d_in[8];
  const float* eW2     = (const float*)d_in[9];
  const float* eB2     = (const float*)d_in[10];
  const float* rW      = (const float*)d_in[11];
  const float* rB      = (const float*)d_in[12];
  const float* subW    = (const float*)d_in[13];
  const float* subB    = (const float*)d_in[14];
  const float* eBias   = (const float*)d_in[15];
  const float* seBias  = (const float*)d_in[16];
  float* out = (float*)d_out;

  // workspace layout (~8.5 MB)
  float* rbuf0   = (float*)d_ws;                       // [N*H] routed slot 0
  float* rbuf1   = rbuf0 + (size_t)NTOK * HDIM;        // [N*H] routed slot 1
  int*   tc0     = (int*)(rbuf1 + (size_t)NTOK * HDIM);
  int*   tc1     = tc0 + NTOK;
  int*   tsi     = tc1 + NTOK;
  float* tw0     = (float*)(tsi + NTOK);
  float* tw1     = tw0 + NTOK;
  float* tsw     = tw1 + NTOK;
  int*   cnt     = (int*)(tsw + NTOK);                 // [36]
  int*   offs    = cnt + NGRP;                         // [37]
  int*   cursor  = offs + NGRP + 1;                    // [36]
  int*   toklist = cursor + NGRP;                      // [3N]
  float* gatelst = (float*)(toklist + 3 * NTOK);       // [3N]

  moe_zero_cnt_kernel<<<1, 64, 0, stream>>>(cnt);
  moe_router_kernel<<<NTOK / 8, 256, 0, stream>>>(
      x, shRW, shRB, subW, subB, rW, rB, eBias, seBias,
      tc0, tw0, tc1, tw1, tsi, tsw, cnt);
  moe_scan_kernel<<<1, 32, 0, stream>>>(cnt, offs, cursor);
  moe_scatter_kernel<<<NTOK / 256, 256, 0, stream>>>(
      tc0, tw0, tc1, tw1, tsi, tsw, cursor, toklist, gatelst);

  size_t shmem = (size_t)2 * MT * HDIM * sizeof(bf16_t) + MT * sizeof(int) + MT * sizeof(float);
  // routed combos: groups 0..31, F = H
  moe_ffn_wmma_kernel<<<dim3(NCOMBO, NTOK / MT), 256, shmem, stream>>>(
      x, eW1, eB1, eW2, eB2, HDIM, 0, 1, offs, toklist, gatelst, rbuf0, rbuf1);
  // shared sub-experts: groups 32..35, F = F_SH; writes `out` directly
  moe_ffn_wmma_kernel<<<dim3(NSUB, NTOK / MT), 256, shmem, stream>>>(
      x, shW1, shB1, shW2, shB2, FSH, NCOMBO, 0, offs, toklist, gatelst, out, out);

  moe_combine_kernel<<<(NTOK * HDIM + 255) / 256, 256, 0, stream>>>(
      out, rbuf0, rbuf1, NTOK * HDIM);
}
